// HierarchicalTriXFFN_51934744543463
// MI455X (gfx1250) — matrix operand order repaired and checked by reading, not compiled, not run on hardware
//
#include <hip/hip_runtime.h>

typedef __attribute__((ext_vector_type(16))) _Float16 v16h;
typedef __attribute__((ext_vector_type(8)))  float    v8f;

#define Dd   2048
#define NTOK 8192
#define Tt   64
#define Hh   128
#define NCl  8
#define TPCc 8
#define EPSF 1e-6f
#define FFN_BLOCKS (NTOK/16 + Tt)   // 576: upper bound on total 16-token chunks

__device__ __forceinline__ float signf(float x) { return (x > 0.f) ? 1.f : ((x < 0.f) ? -1.f : 0.f); }

// ---------------------------------------------------------------------------
// Pack sign(src) into WMMA 16x16x32 f16 B-fragment order.
// src layout [T][N][K] (N = output units, K = reduction dim).
// Fragment: base = (((t*(K/32)+kb)*(N/16)+nb)*32 + lane)*16 halves.
// Element j of lane l = sign(src[t][ nb*16+(l&15) ][ kb*32 + 16*(l>>4) + j ]).
// ---------------------------------------------------------------------------
__global__ void pack_sign_kernel(const float* __restrict__ src, _Float16* __restrict__ dst,
                                 int Ndim, int Kdim) {
  long gidx = (long)blockIdx.x * blockDim.x + threadIdx.x;
  long tot  = (long)Tt * (Kdim / 32) * (Ndim / 16) * 32;
  if (gidx >= tot) return;
  int  lane = (int)(gidx & 31);
  long rest = gidx >> 5;
  int  nb = (int)(rest % (Ndim / 16)); rest /= (Ndim / 16);
  int  kb = (int)(rest % (Kdim / 32));
  int  t  = (int)(rest / (Kdim / 32));
  int  hh = lane >> 4;
  int  n  = nb * 16 + (lane & 15);
  const float* s = src + ((long)t * Ndim + n) * Kdim + kb * 32 + 16 * hh;
  v16h o;
#pragma unroll
  for (int j = 0; j < 16; ++j) o[j] = (_Float16)signf(s[j]);
  *(v16h*)(dst + gidx * 16) = o;
}

// tile_sig[t][d] = sign( sum_h sign(up_w[t][h][d]) )
__global__ void tile_sig_kernel(const float* __restrict__ up_w, float* __restrict__ tsig) {
  int idx = blockIdx.x * blockDim.x + threadIdx.x;
  if (idx >= Tt * Dd) return;
  int t = idx / Dd, d = idx % Dd;
  float s = 0.f;
  for (int h = 0; h < Hh; ++h) s += signf(up_w[((long)t * Hh + h) * Dd + d]);
  tsig[idx] = signf(s);
}

// cluster_sig[c][d] = sign( mean_i tile_sig[c*TPC+i][d] )
__global__ void cluster_sig_kernel(const float* __restrict__ tsig, float* __restrict__ csig) {
  int idx = blockIdx.x * blockDim.x + threadIdx.x;
  if (idx >= NCl * Dd) return;
  int c = idx / Dd, d = idx % Dd;
  float s = 0.f;
  for (int i = 0; i < TPCc; ++i) s += tsig[(long)(c * TPCc + i) * Dd + d];
  csig[idx] = signf(s);
}

__global__ void init_kernel(int* __restrict__ counts, int* __restrict__ cursor) {
  int i = threadIdx.x;
  if (i < Tt) { counts[i] = 0; cursor[i] = 0; }
}

// One block per token: RMSNorm -> xn (f16), 2-level routing, tile histogram.
__global__ __launch_bounds__(256) void rmsnorm_route_kernel(
    const float* __restrict__ x, const float* __restrict__ nw,
    const float* __restrict__ csig, const float* __restrict__ tsig,
    _Float16* __restrict__ xnh, int* __restrict__ tile_idx, int* __restrict__ counts) {
  __shared__ _Float16 xl[Dd];
  __shared__ float red[8];
  __shared__ float sc[8];
  __shared__ int cbest;
  int n = blockIdx.x, tid = threadIdx.x;
  int lane = tid & 31, wv = tid >> 5;
  const float* xr = x + (long)n * Dd;

  float ss = 0.f;
  for (int d = tid; d < Dd; d += 256) { float v = xr[d]; ss += v * v; }
  for (int o = 16; o; o >>= 1) ss += __shfl_xor(ss, o, 32);
  if (!lane) red[wv] = ss;
  __syncthreads();
  if (tid == 0) {
    float s = 0.f;
    for (int i = 0; i < 8; ++i) s += red[i];
    red[0] = rsqrtf(s / (float)Dd + EPSF);
  }
  __syncthreads();
  float inv = red[0];
  for (int d = tid; d < Dd; d += 256) {
    _Float16 hv = (_Float16)(xr[d] * inv * nw[d]);
    xl[d] = hv;
    xnh[(long)n * Dd + d] = hv;
  }
  __syncthreads();
  { // wave wv scores cluster wv
    const float* cs = csig + (long)wv * Dd;
    float p = 0.f;
    for (int d = lane; d < Dd; d += 32) p += (float)xl[d] * cs[d];
    for (int o = 16; o; o >>= 1) p += __shfl_xor(p, o, 32);
    if (!lane) sc[wv] = p;
  }
  __syncthreads();
  if (tid == 0) {
    int b = 0; float m = sc[0];
    for (int i = 1; i < 8; ++i) if (sc[i] > m) { m = sc[i]; b = i; }
    cbest = b;
  }
  __syncthreads();
  { // wave wv scores tile (cbest*TPC + wv)
    const float* ts = tsig + (long)(cbest * TPCc + wv) * Dd;
    float p = 0.f;
    for (int d = lane; d < Dd; d += 32) p += (float)xl[d] * ts[d];
    for (int o = 16; o; o >>= 1) p += __shfl_xor(p, o, 32);
    if (!lane) sc[wv] = p;
  }
  __syncthreads();
  if (tid == 0) {
    int b = 0; float m = sc[0];
    for (int i = 1; i < 8; ++i) if (sc[i] > m) { m = sc[i]; b = i; }
    int tl = cbest * TPCc + b;
    tile_idx[n] = tl;
    atomicAdd(&counts[tl], 1);
  }
}

__global__ void scan_kernel(const int* __restrict__ counts, int* __restrict__ tokOff,
                            int* __restrict__ chunkOff, int* __restrict__ totalChunks) {
  if (threadIdx.x == 0 && blockIdx.x == 0) {
    int a = 0, c = 0;
    for (int t = 0; t < Tt; ++t) {
      tokOff[t] = a; chunkOff[t] = c;
      a += counts[t]; c += (counts[t] + 15) >> 4;
    }
    tokOff[Tt] = a; chunkOff[Tt] = c; *totalChunks = c;
  }
}

__global__ void scatter_kernel(const int* __restrict__ tile_idx, const int* __restrict__ tokOff,
                               int* __restrict__ cursor, int* __restrict__ token_list) {
  int n = blockIdx.x * blockDim.x + threadIdx.x;
  if (n >= NTOK) return;
  int t = tile_idx[n];
  int p = atomicAdd(&cursor[t], 1);
  token_list[tokOff[t] + p] = n;
}

// ---------------------------------------------------------------------------
// Grouped FFN: one block = (tile, chunk of <=16 tokens). 8 waves.
// Up: each wave owns one 16-col H-tile; K=2048 streamed via double-buffered
//     512-wide LDS panels so A staging overlaps WMMA.
// Down: hidden A-fragments hoisted to registers; pure B-stream -> WMMA.
// WMMA f32_16x16x32_f16; EXEC all-ones (uniform control flow around WMMA).
// ---------------------------------------------------------------------------
__global__ __launch_bounds__(256) void ffn_kernel(
    const _Float16* __restrict__ xnh, const _Float16* __restrict__ upPk,
    const _Float16* __restrict__ dnPk, const float* __restrict__ up_scale,
    const float* __restrict__ down_scale, const float* __restrict__ out_scale,
    const float* __restrict__ x, const int* __restrict__ token_list,
    const int* __restrict__ tokOff, const int* __restrict__ chunkOff,
    const int* __restrict__ totalChunks, float* __restrict__ out) {
  __shared__ int tokens[16];
  __shared__ _Float16 As[2][16][520];  // double-buffered 512-wide K panel + pad
  __shared__ _Float16 Hs[16][136];     // hidden 16x128 + pad

  int b = blockIdx.x;
  if (b >= *totalChunks) return;       // whole block exits uniformly

  int t = 0;
  while (t < Tt - 1 && b >= chunkOff[t + 1]) ++t;
  int chunk = b - chunkOff[t];
  int cnt   = tokOff[t + 1] - tokOff[t];
  int rows  = cnt - chunk * 16; if (rows > 16) rows = 16;

  int tid = threadIdx.x;
  if (tid < 16) tokens[tid] = (tid < rows) ? token_list[tokOff[t] + chunk * 16 + tid] : -1;
  int lane = tid & 31, wv = tid >> 5, hh = lane >> 4, mrow = lane & 15;
  __syncthreads();  // tokens[] visible to all stagers

  auto stage = [&](int p, int bufi) {
    for (int idx = tid; idx < 16 * 64; idx += 256) {
      int r = idx >> 6, c = idx & 63;
      int tk = tokens[r];
      uint4 val = make_uint4(0u, 0u, 0u, 0u);
      if (tk >= 0) val = *(const uint4*)(xnh + (long)tk * Dd + p * 512 + c * 8);
      *(uint4*)&As[bufi][r][c * 8] = val;
    }
  };

  stage(0, 0);

  v8f accU;
#pragma unroll
  for (int i = 0; i < 8; ++i) accU[i] = 0.f;

  for (int p = 0; p < 4; ++p) {
    __syncthreads();                    // panel p staged; prior reads drained
    if (p + 1 < 4) stage(p + 1, (p + 1) & 1);  // overlap next panel with WMMAs
    int bufi = p & 1;
#pragma unroll 4
    for (int kk = 0; kk < 512; kk += 32) {
      v16h a, bm;
      const _Float16* p0 = &As[bufi][mrow][kk + hh * 16];
      const _Float16* p1 = &As[bufi][mrow][kk + hh * 16 + 8];
#pragma unroll
      for (int j = 0; j < 8; ++j) { a[j] = p0[j]; a[j + 8] = p1[j]; }
      long kbG = (long)((p * 512 + kk) >> 5);
      bm = *(const v16h*)(upPk + ((((long)t * (Dd / 32) + kbG) * (Hh / 16) + wv) * 32 + lane) * 16);
      accU = __builtin_amdgcn_wmma_f32_16x16x32_f16(false, a, false, bm, (short)0, accU, false, false);
    }
  }

  // ReLU + up_scale -> LDS hidden (f16)
  {
    int nloc = wv * 16 + (lane & 15);
    float us = up_scale[t * Hh + nloc];
#pragma unroll
    for (int v = 0; v < 8; ++v) {
      int m = v + 8 * hh;
      float hv = accU[v] * us;
      Hs[m][nloc] = (_Float16)(hv > 0.f ? hv : 0.f);
    }
  }
  __syncthreads();

  // Hoist the 4 down-GEMM A-fragments (invariant across the 16 N-tiles)
  v16h aD[4];
#pragma unroll
  for (int kb = 0; kb < 4; ++kb) {
    const _Float16* p0 = &Hs[mrow][kb * 32 + hh * 16];
    const _Float16* p1 = &Hs[mrow][kb * 32 + hh * 16 + 8];
#pragma unroll
    for (int j = 0; j < 8; ++j) { aD[kb][j] = p0[j]; aD[kb][j + 8] = p1[j]; }
  }

  float osc = out_scale[t];
  const _Float16* dnBase = dnPk + (long)t * (Hh / 32) * (Dd / 16) * 512 + (long)lane * 16;
#pragma unroll 2
  for (int i = 0; i < 16; ++i) {
    int nb = wv * 16 + i;
    v8f acc;
#pragma unroll
    for (int q = 0; q < 8; ++q) acc[q] = 0.f;
#pragma unroll
    for (int kb = 0; kb < 4; ++kb) {
      v16h bm = *(const v16h*)(dnBase + ((long)kb * (Dd / 16) + nb) * 512);
      acc = __builtin_amdgcn_wmma_f32_16x16x32_f16(false, aD[kb], false, bm, (short)0, acc, false, false);
    }
    int nglob = nb * 16 + (lane & 15);
    float dsc = down_scale[(long)t * Dd + nglob] * osc;
#pragma unroll
    for (int v = 0; v < 8; ++v) {
      int m = v + 8 * hh;
      if (m < rows) {
        int tk = tokens[m];
        out[(long)tk * Dd + nglob] = x[(long)tk * Dd + nglob] + acc[v] * dsc;
      }
    }
  }
}

extern "C" void kernel_launch(void* const* d_in, const int* in_sizes, int n_in,
                              void* d_out, int out_size, void* d_ws, size_t ws_size,
                              hipStream_t stream) {
  (void)in_sizes; (void)n_in; (void)out_size; (void)ws_size;
  const float* x        = (const float*)d_in[0];
  const float* up_w     = (const float*)d_in[1];
  const float* down_w   = (const float*)d_in[2];
  const float* up_sc    = (const float*)d_in[3];
  const float* down_sc  = (const float*)d_in[4];
  const float* out_sc   = (const float*)d_in[5];
  const float* norm_w   = (const float*)d_in[6];
  float* out = (float*)d_out;

  // workspace carve-up (256B aligned)
  char* ws = (char*)d_ws;
  size_t off = 0;
  auto take = [&](size_t bytes) { char* p = ws + off; off = (off + bytes + 255) & ~(size_t)255; return p; };
  _Float16* xnh   = (_Float16*)take((size_t)NTOK * Dd * 2);
  _Float16* upPk  = (_Float16*)take((size_t)Tt * (Dd / 32) * (Hh / 16) * 512 * 2);
  _Float16* dnPk  = (_Float16*)take((size_t)Tt * (Hh / 32) * (Dd / 16) * 512 * 2);
  float*    tsig  = (float*)take((size_t)Tt * Dd * 4);
  float*    csig  = (float*)take((size_t)NCl * Dd * 4);
  int*      tidx  = (int*)take((size_t)NTOK * 4);
  int*      counts= (int*)take(Tt * 4);
  int*      tokOff= (int*)take((Tt + 1) * 4);
  int*      chkOff= (int*)take((Tt + 1) * 4);
  int*      cursor= (int*)take(Tt * 4);
  int*      tlist = (int*)take((size_t)NTOK * 4);
  int*      totCh = (int*)take(4);

  // 1) pack sign weights into WMMA fragment order (up: N=H,K=D; down: N=D,K=H)
  {
    long thr = (long)Tt * (Dd / 32) * (Hh / 16) * 32;  // == down's total too
    int blks = (int)((thr + 255) / 256);
    pack_sign_kernel<<<blks, 256, 0, stream>>>(up_w, upPk, Hh, Dd);
    pack_sign_kernel<<<blks, 256, 0, stream>>>(down_w, dnPk, Dd, Hh);
  }
  // 2) signatures
  tile_sig_kernel<<<(Tt * Dd + 255) / 256, 256, 0, stream>>>(up_w, tsig);
  cluster_sig_kernel<<<(NCl * Dd + 255) / 256, 256, 0, stream>>>(tsig, csig);
  // 3) routing
  init_kernel<<<1, 128, 0, stream>>>(counts, cursor);
  rmsnorm_route_kernel<<<NTOK, 256, 0, stream>>>(x, norm_w, csig, tsig, xnh, tidx, counts);
  // 4) bucket tokens by tile
  scan_kernel<<<1, 64, 0, stream>>>(counts, tokOff, chkOff, totCh);
  scatter_kernel<<<(NTOK + 255) / 256, 256, 0, stream>>>(tidx, tokOff, cursor, tlist);
  // 5) grouped WMMA FFN + residual
  ffn_kernel<<<FFN_BLOCKS, 256, 0, stream>>>(xnh, upPk, dnPk, up_sc, down_sc, out_sc,
                                             x, tlist, tokOff, chkOff, totCh, out);
}
